// AutoCorrelation_29996051595366
// MI455X (gfx1250) — matrix operand order repaired and checked by reading, not compiled, not run on hardware
//
#include <hip/hip_runtime.h>
#include <math.h>

// ---------------------------------------------------------------------------
// AutoCorrelation (Autoformer) on gfx1250:
//   corr[d] = sum_t Q[t]*K[(t-d) mod L]  recast as GEMM on bf16 WMMA pipes:
//     A[m,t] = Q[(t + dbase + 16m) mod L], B[t,n] = K[(t-n) mod L]
//     => D[m,n] = corr[dbase + 16m + n]
//   then per-series top-15 + softmax + circular gather from V.
// Shapes fixed by the reference: B=32,H=8,L=2048,E=64, top_k=15.
//
// LDS per wave (ushort units):
//   [0    .. 4095] : Q series duplicated (Q[l], then Q[l] again) -> unwrapped
//                    A indices (max 4079) need no modulo.
//   [4096 .. 6159] : K series with 16-entry front pad:
//                    slot 4096+j == K[(j-16) mod 2048]  -> B indices need no
//                    modulo (tb+i+16 in [1,2063]).
//   region reused later as 2048 f32 corr values (8192 B <= 12352 B).
// ---------------------------------------------------------------------------

typedef __attribute__((ext_vector_type(16))) __bf16 v16bf;
typedef __attribute__((ext_vector_type(8)))  float  v8f;

#define L_LEN 2048
#define E_DIM 64
#define TOPK  15
#define NEG_BIG -3.402823466e38f
#define WREG  6176   // ushorts per wave region (16B-aligned: 12352 bytes)

union FragAB {
  v16bf v;
  unsigned short u[16];
  uint4 q[2];
};

__device__ __forceinline__ unsigned short f2bf(float f) {
  // round-to-nearest-even float32 -> bf16 (bit trick)
  unsigned u = __float_as_uint(f);
  unsigned r = u + 0x7FFFu + ((u >> 16) & 1u);
  return (unsigned short)(r >> 16);
}

__global__ __launch_bounds__(128)
void autocorr_wmma_kernel(const float* __restrict__ Q,
                          const float* __restrict__ K,
                          const float* __restrict__ V,
                          float* __restrict__ Out) {
  __shared__ alignas(16) unsigned short sLds[4 * WREG];   // 49408 bytes

  const int bh    = blockIdx.x >> 4;   // B*H = 256 pairs
  const int etile = blockIdx.x & 15;   // 16 tiles of 4 e-channels
  const int e0    = etile * 4;
  const int tid   = threadIdx.x;
  const int w     = tid >> 5;          // wave id: one e-channel per wave
  const int lane  = tid & 31;
  const int hl    = lane >> 4;         // lane half (K-subset select)
  const int nl    = lane & 15;         // A: M row, B: N col, D: N col

  const float* Qb = Q + (size_t)bh * L_LEN * E_DIM;
  const float* Kb = K + (size_t)bh * L_LEN * E_DIM;
  const float* Vb = V + (size_t)bh * L_LEN * E_DIM;
  float*       Ob = Out + (size_t)bh * L_LEN * E_DIM;

  // ---- Stage Q (duplicated) and K (front-padded) as bf16 into LDS ----
  for (int it = 0; it < 16; ++it) {
    int l = it * 128 + tid;                       // 0..2047, no masking
    float4 q4 = *(const float4*)(Qb + (size_t)l * E_DIM + e0);
    float4 k4 = *(const float4*)(Kb + (size_t)l * E_DIM + e0);
    unsigned short qb[4] = { f2bf(q4.x), f2bf(q4.y), f2bf(q4.z), f2bf(q4.w) };
    unsigned short kb[4] = { f2bf(k4.x), f2bf(k4.y), f2bf(k4.z), f2bf(k4.w) };
#pragma unroll
    for (int j = 0; j < 4; ++j) {
      unsigned short* reg = &sLds[j * WREG];
      reg[l]        = qb[j];            // Q copy 0
      reg[2048 + l] = qb[j];            // Q copy 1 (unwrapped indices)
      reg[4096 + 16 + l] = kb[j];       // K main body
      if (l >= 2033) reg[4096 + (l - 2032)] = kb[j];  // K front pad
    }
  }
  __syncthreads();

  unsigned short* sQ = &sLds[w * WREG];          // 4096 entries (dup)
  unsigned short* sK = &sLds[w * WREG + 4096];   // 2064 entries (padded)
  const int e = e0 + w;

  // Per-lane invariant bases (linear, no modulo in the hot loop).
  const unsigned short* aBase = sQ + (hl * 8 + nl * 16);   // + kc*32 + t*256 (+16)
  const unsigned short* kBase = sK + (16 + hl * 8 - nl);   // + kc*32 + i (+8 for i>=8)

  // ---- GEMM: 8 output tiles of 16x16 delays, K-contraction = 2048 ----
  v8f acc[8];
#pragma unroll
  for (int t = 0; t < 8; ++t) acc[t] = (v8f){0.f,0.f,0.f,0.f,0.f,0.f,0.f,0.f};

#pragma unroll 2
  for (int kc = 0; kc < 64; ++kc) {
    // B fragment: b.u[i] = K[(kc*32 + hl*8 + i + (i>=8?8:0) - nl) mod L], col n = nl
    FragAB bfr;
    const unsigned short* kp = kBase + kc * 32;
#pragma unroll
    for (int i = 0; i < 16; ++i)
      bfr.u[i] = kp[i + (i >= 8 ? 8 : 0)];

    const unsigned short* ap = aBase + kc * 32;
#pragma unroll
    for (int t = 0; t < 8; ++t) {
      // A fragment: two 16B-aligned LDS loads at immediate offsets.
      FragAB afr;
      afr.q[0] = *(const uint4*)(ap + t * 256);
      afr.q[1] = *(const uint4*)(ap + t * 256 + 16);
      acc[t] = __builtin_amdgcn_wmma_f32_16x16x32_bf16(
          false, afr.v, false, bfr.v, (short)0, acc[t], false, false);
    }
  }

  // ---- Dump corr into this wave's (now dead) LDS region ----
  float* sC = (float*)sQ;   // 2048 f32 fits in the 12352-byte region
#pragma unroll
  for (int t = 0; t < 8; ++t)
#pragma unroll
    for (int r = 0; r < 8; ++r)
      sC[t * 256 + (r + hl * 8) * 16 + nl] = acc[t][r];
  // same-wave DS ops are in-order: stores visible to the loads below

  // ---- top-15 delays via 15 rounds of wave-wide argmax ----
  float wv[TOPK];
  int   wd[TOPK];
#pragma unroll
  for (int kk = 0; kk < TOPK; ++kk) {
    float best = NEG_BIG;
    int   bidx = 0;
    for (int j = lane; j < L_LEN; j += 32) {
      float v = sC[j];
      if (v > best) { best = v; bidx = j; }
    }
#pragma unroll
    for (int off = 16; off > 0; off >>= 1) {
      float ov = __shfl_xor(best, off, 32);
      int   oi = __shfl_xor(bidx, off, 32);
      if (ov > best || (ov == best && oi < bidx)) { best = ov; bidx = oi; }
    }
    wv[kk] = best;
    wd[kk] = bidx;
    sC[bidx] = NEG_BIG;   // all lanes, same addr, same value
  }

  // ---- softmax over the 15 weights (descending, wv[0] is the max) ----
  float pw[TOPK];
  float s = 0.f;
#pragma unroll
  for (int kk = 0; kk < TOPK; ++kk) {
    pw[kk] = __expf(wv[kk] - wv[0]);
    s += pw[kk];
  }
  const float inv = 1.f / s;

  // ---- out[l] = sum_i w_i * V[(l + d_i) mod L]  (gather from L2) ----
  const float* Ve = Vb + e;
  float*       Oe = Ob + e;
  for (int l = lane; l < L_LEN; l += 32) {
    float o = 0.f;
#pragma unroll
    for (int kk = 0; kk < TOPK; ++kk)
      o += pw[kk] * Ve[(size_t)((l + wd[kk]) & (L_LEN - 1)) * E_DIM];
    Oe[(size_t)l * E_DIM] = o * inv;
  }
}

extern "C" void kernel_launch(void* const* d_in, const int* in_sizes, int n_in,
                              void* d_out, int out_size, void* d_ws, size_t ws_size,
                              hipStream_t stream) {
  (void)n_in; (void)out_size; (void)d_ws; (void)ws_size;
  const float* Q = (const float*)d_in[0];
  const float* K = (const float*)d_in[1];
  const float* V = (const float*)d_in[2];
  float* Out = (float*)d_out;

  const int BH = in_sizes[0] / (L_LEN * E_DIM);   // = 32*8 = 256
  dim3 grid(BH * (E_DIM / 4));                    // 4096 blocks
  dim3 block(128);                                // 4 waves, 1 e-series each
  autocorr_wmma_kernel<<<grid, block, 0, stream>>>(Q, K, V, Out);
}
